// Transformer_76553497084444
// MI455X (gfx1250) — matrix operand order repaired and checked by reading, not compiled, not run on hardware
//
#include <hip/hip_runtime.h>

// ---------------------------------------------------------------------------
// GPT-style transformer forward on MI455X (gfx1250), bf16 WMMA everywhere.
// L=6, B=4, T=1024, E=512, H=8, D=64, V=256
// GEMM: 128x64 block tile, double-buffered LDS, A-tile via TDM
// (tensor_load_to_lds + s_wait_tensorcnt), B-tile via sync transpose stage.
// ---------------------------------------------------------------------------

#define DEVI __device__ __forceinline__

typedef unsigned short u16;
typedef __attribute__((ext_vector_type(4)))  unsigned int u32x4;
typedef __attribute__((ext_vector_type(4)))  int          s32x4;
typedef __attribute__((ext_vector_type(8)))  int          s32x8;
typedef __attribute__((ext_vector_type(16))) __bf16       v16bf;
typedef __attribute__((ext_vector_type(8)))  float        v8f;

union Frag {
    v16bf v;
    u16   u[16];
    u32x4 q[2];
};

enum : int { L_ = 6, B_ = 4, T_ = 1024, E_ = 512, H_ = 8, D_ = 64, V_ = 256 };
enum : int { E3_ = 3 * E_, E4_ = 4 * E_, BT_ = B_ * T_ };
#define ATT_SCALE 0.125f  /* 1/sqrt(64) */

DEVI u16 f2bf(float f) {  // round-to-nearest-even f32 -> bf16
    unsigned int u = __float_as_uint(f);
    return (u16)((u + 0x7FFFu + ((u >> 16) & 1u)) >> 16);
}

DEVI v8f wmma_bf16(v16bf a, v16bf b, v8f c) {
    return __builtin_amdgcn_wmma_f32_16x16x32_bf16(false, a, false, b, (short)0, c,
                                                   false, false);
}

// ---------------------------------------------------------------------------
// Tensor Data Mover: 2D tile load global->LDS (bf16 elements).
// Builds the 4-group D# (CDNA5 ISA) in SGPRs and issues
// tensor_load_to_lds. Wave-uniform arguments required. Tracked by TENSORcnt.
//   tile_w elems per row (mult of 2 for 4B padding rule), tile_h rows,
//   row_stride = tensor_dim0_stride (elems), tensor_w/h for OOB bounds.
// ---------------------------------------------------------------------------
DEVI void tdm_load_tile_2d(unsigned lds_addr, const u16* gptr,
                           int tile_w, int tile_h,
                           int row_stride, int tensor_w, int tensor_h) {
    unsigned long long ga = (unsigned long long)(const void*)gptr;
    s32x4 g0;
    s32x8 g1;
    s32x4 gz = {0, 0, 0, 0};
    g0[0] = 1;                                           // count=1, user D#
    g0[1] = (int)lds_addr;                               // lds_addr[31:0]
    g0[2] = (int)(unsigned)(ga & 0xFFFFFFFFu);           // global_addr[31:0]
    g0[3] = (int)((unsigned)((ga >> 32) & 0x01FFFFFFu) | (2u << 30));  // type=2
    g1[0] = (1 << 16);                                   // data_size=2B, mask=0
    g1[1] = (int)(((unsigned)tensor_w & 0xFFFFu) << 16); // tensor_dim0[15:0]
    g1[2] = (int)(((unsigned)tensor_w >> 16) |
                  (((unsigned)tensor_h & 0xFFFFu) << 16));
    g1[3] = (int)(((unsigned)tensor_h >> 16) |
                  (((unsigned)tile_w & 0xFFFFu) << 16)); // tile_dim0
    g1[4] = (int)((unsigned)tile_h & 0xFFFFu);           // tile_dim1, tile_dim2=0
    g1[5] = row_stride;                                  // tensor_dim0_stride lo
    g1[6] = 0;
    g1[7] = 0;
    asm volatile("tensor_load_to_lds %0, %1, %2, %3"
                 :: "s"(g0), "s"(g1), "s"(gz), "s"(gz)
                 : "memory");
}

// ---------------------------------------------------------------------------
// f32 -> bf16 bulk convert (n multiple of 4)
// ---------------------------------------------------------------------------
__global__ void __launch_bounds__(256) k_cvt_bf16(const float* __restrict__ in,
                                                  u16* __restrict__ out, long long n) {
    long long i = ((long long)blockIdx.x * 256 + threadIdx.x) * 4;
    if (i >= n) return;
    float4 f = *(const float4*)(in + i);
    out[i + 0] = f2bf(f.x);
    out[i + 1] = f2bf(f.y);
    out[i + 2] = f2bf(f.z);
    out[i + 3] = f2bf(f.w);
}

// ---------------------------------------------------------------------------
// Embedding: x[b,t,:] = wte[idx[b,t],:] + wpe[t,:]
// ---------------------------------------------------------------------------
__global__ void __launch_bounds__(256) k_embed(const int* __restrict__ idx,
                                               const float* __restrict__ wte,
                                               const float* __restrict__ wpe,
                                               float* __restrict__ x) {
    int i  = blockIdx.x * 256 + threadIdx.x;   // over B*T*E
    int e  = i % E_;
    int bt = i / E_;
    int tt = bt % T_;
    x[i] = wte[(size_t)idx[bt] * E_ + e] + wpe[(size_t)tt * E_ + e];
}

// ---------------------------------------------------------------------------
// LayerNorm over E=512, one block per row, bf16 output (feeds WMMA GEMM)
// ---------------------------------------------------------------------------
__global__ void __launch_bounds__(256) k_layernorm(const float* __restrict__ x,
                                                   const float* __restrict__ g,
                                                   const float* __restrict__ b,
                                                   u16* __restrict__ out) {
    __shared__ float red[256];
    int row = blockIdx.x;
    int t   = threadIdx.x;
    const float* xr = x + (size_t)row * E_;
    float v0 = xr[t], v1 = xr[t + 256];

    red[t] = v0 + v1;
    __syncthreads();
    for (int o = 128; o > 0; o >>= 1) {
        if (t < o) red[t] += red[t + o];
        __syncthreads();
    }
    float mean = red[0] * (1.0f / E_);
    __syncthreads();

    float d0 = v0 - mean, d1 = v1 - mean;
    red[t] = d0 * d0 + d1 * d1;
    __syncthreads();
    for (int o = 128; o > 0; o >>= 1) {
        if (t < o) red[t] += red[t + o];
        __syncthreads();
    }
    float inv = rsqrtf(red[0] * (1.0f / E_) + 1e-5f);

    out[(size_t)row * E_ + t]       = f2bf(d0 * inv * g[t] + b[t]);
    out[(size_t)row * E_ + t + 256] = f2bf(d1 * inv * g[t + 256] + b[t + 256]);
}

// ---------------------------------------------------------------------------
// Tiled WMMA GEMM: C[M,N] = A[M,K](bf16) * B[K,N](bf16) + bias
// 128x64 block tile, 8 waves; wave w owns rows [w*16, w*16+16) and all four
// 16x16 n-tiles (one A fragment amortized over 4 WMMAs per K-step).
// Double-buffered LDS, ONE barrier per K-step:
//   A tile (128x32) : TDM tensor_load_to_lds, awaited via s_wait_tensorcnt
//   B tile (32x64)  : cooperative 128-bit loads + in-VGPR transpose to [n][k]
// act: 0=none, 1=tanh-GELU. Cf / Cbf optional outputs.
// ---------------------------------------------------------------------------
__global__ void __launch_bounds__(256) k_gemm_bf16(const u16* __restrict__ A,
                                                   const u16* __restrict__ B,
                                                   const float* __restrict__ bias,
                                                   float* __restrict__ Cf,
                                                   u16* __restrict__ Cbf,
                                                   int M, int N, int K, int act) {
    __shared__ __align__(16) u16 As[2][128 * 32];  // [m][k], TDM-filled
    __shared__ __align__(16) u16 Bs[2][64 * 32];   // [n][k] transposed stage

    int n0 = blockIdx.x * 64, m0 = blockIdx.y * 128;
    int t = threadIdx.x, lane = t & 31, wave = t >> 5;

    v8f acc[4] = {{}, {}, {}, {}};

    int mA = wave * 16 + (lane & 15);
    int b0 = (lane < 16) ? 0 : 8;   // 16-bit A lane layout: two 8-elem runs
    int nb = lane & 15;
    int kb = (lane >> 4) * 8;       // B lane layout: two 8-elem runs

    unsigned ldsA0 = (unsigned)(unsigned long long)(const void*)&As[0][0];
    unsigned ldsA1 = (unsigned)(unsigned long long)(const void*)&As[1][0];

    // prologue: TDM load of first A tile
    if (t < 32) {
        tdm_load_tile_2d(ldsA0, A + (size_t)m0 * K, 32, 128, K, K, M);
    }

    int NB = K >> 5;
    for (int i = 0; i < NB; ++i) {
        int k0  = i << 5;
        int cur = i & 1;

        // stage B tile (32x64) transposed into Bs[cur]
        {
            int kr = t >> 3, ng = (t & 7) * 8;
            const u16* bsrc = B + (size_t)(k0 + kr) * N + n0 + ng;
            u32x4 val = *(const u32x4*)bsrc;
            if (k0 + 32 < K)  // prefetch next B tile (global_prefetch_b8)
                __builtin_prefetch((const void*)(bsrc + (size_t)32 * N), 0, 3);
            u16 tmp[8];
            *(u32x4*)tmp = val;
#pragma unroll
            for (int e = 0; e < 8; ++e) Bs[cur][(ng + e) * 32 + kr] = tmp[e];
        }

        if (t < 32) __builtin_amdgcn_s_wait_tensorcnt(0);  // A[cur] in LDS
        __syncthreads();

        // issue TDM for the next A tile into the other buffer
        if (i + 1 < NB && t < 32) {
            tdm_load_tile_2d(cur ? ldsA0 : ldsA1,
                             A + (size_t)m0 * K + (k0 + 32), 32, 128, K, K, M);
        }

        const u16* Asc = &As[cur][0];
        const u16* Bsc = &Bs[cur][0];

        Frag a;
        a.q[0] = *(const u32x4*)(Asc + mA * 32 + b0);
        a.q[1] = *(const u32x4*)(Asc + mA * 32 + b0 + 16);
#pragma unroll
        for (int nt = 0; nt < 4; ++nt) {
            Frag bf;
            const u16* bp = Bsc + (nt * 16 + nb) * 32 + kb;
            bf.q[0] = *(const u32x4*)bp;
            bf.q[1] = *(const u32x4*)(bp + 16);
            acc[nt] = wmma_bf16(a.v, bf.v, acc[nt]);
        }
        // no trailing barrier: next iteration stages the opposite buffers;
        // the single barrier above orders reuse two iterations apart.
    }

#pragma unroll
    for (int nt = 0; nt < 4; ++nt) {
        v8f c = acc[nt];
        int n = n0 + nt * 16 + nb;
        float bv = bias ? bias[n] : 0.0f;
#pragma unroll
        for (int r = 0; r < 8; ++r) {
            int m = m0 + wave * 16 + r + ((lane >> 4) * 8);
            float v = c[r] + bv;
            if (act == 1) {
                float xx = v;
                v = 0.5f * xx *
                    (1.0f + tanhf(0.7978845608f * (xx + 0.044715f * xx * xx * xx)));
            }
            size_t o = (size_t)m * N + n;
            if (Cf)  Cf[o]  = v;
            if (Cbf) Cbf[o] = f2bf(v);
        }
    }
}

// ---------------------------------------------------------------------------
// Split qkv[B,T,3E] (f32) -> q,k bf16 [B,H,T,D], v bf16 transposed [B,H,D,T]
// ---------------------------------------------------------------------------
__global__ void __launch_bounds__(256) k_qkv_split(const float* __restrict__ qkv,
                                                   u16* __restrict__ q,
                                                   u16* __restrict__ k,
                                                   u16* __restrict__ vt) {
    int i  = blockIdx.x * 256 + threadIdx.x;  // over B*T*E
    int e  = i % E_;
    int bt = i / E_;
    int b  = bt / T_;
    int tt = bt % T_;
    int h  = e / D_;
    int d  = e % D_;
    const float* base = qkv + (size_t)bt * E3_;
    size_t bh = (size_t)(b * H_ + h);
    q[(bh * T_ + tt) * D_ + d]  = f2bf(base[e]);
    k[(bh * T_ + tt) * D_ + d]  = f2bf(base[E_ + e]);
    vt[(bh * D_ + d) * T_ + tt] = f2bf(base[2 * E_ + e]);
}

// ---------------------------------------------------------------------------
// Attention: one block per (b, h, 16-query tile).
//   Phase 1: S = scale * Q Kt  (WMMA, 8 waves over 64 key tiles) -> LDS 16x1024
//   Phase 2: causal softmax in LDS; post-softmax P streamed to att output
//   Phase 3: Y = P V (WMMA, 4 waves over D tiles), Y written as bf16
// Dynamic LDS: 16*1024 f32 (64KB, fits the 320KB WGP LDS) + reductions.
// ---------------------------------------------------------------------------
__global__ void __launch_bounds__(256) k_attention(const u16* __restrict__ Q,
                                                   const u16* __restrict__ Km,
                                                   const u16* __restrict__ Vt,
                                                   float* __restrict__ att,
                                                   u16* __restrict__ Ybf) {
    extern __shared__ __align__(16) char smem[];
    float* S    = (float*)smem;                       // [16][1024]
    float* red  = (float*)(smem + 16 * T_ * 4);       // [16][16]
    float* rowv = red + 256;                          // [16]

    int qb = blockIdx.x * 16, h = blockIdx.y, b = blockIdx.z;
    int t = threadIdx.x, lane = t & 31, wave = t >> 5;
    size_t bh = (size_t)(b * H_ + h);

    int mrow = lane & 15;
    int b0   = (lane < 16) ? 0 : 8;
    int nb   = lane & 15;
    int kb   = (lane >> 4) * 8;

    // ---- Phase 1: S = scale * Q Kt -------------------------------------
    {
        const u16* qrow = Q + (bh * T_ + qb + mrow) * D_;
        Frag aq0, aq1;
        aq0.q[0] = *(const u32x4*)(qrow + b0);
        aq0.q[1] = *(const u32x4*)(qrow + b0 + 16);
        aq1.q[0] = *(const u32x4*)(qrow + 32 + b0);
        aq1.q[1] = *(const u32x4*)(qrow + 32 + b0 + 16);

        for (int kt = wave; kt < T_ / 16; kt += 8) {
            const u16* krow = Km + (bh * T_ + kt * 16 + nb) * D_;
            Frag bk0, bk1;
            bk0.q[0] = *(const u32x4*)(krow + kb);
            bk0.q[1] = *(const u32x4*)(krow + kb + 16);
            bk1.q[0] = *(const u32x4*)(krow + 32 + kb);
            bk1.q[1] = *(const u32x4*)(krow + 32 + kb + 16);
            v8f c = {};
            c = wmma_bf16(aq0.v, bk0.v, c);
            c = wmma_bf16(aq1.v, bk1.v, c);
#pragma unroll
            for (int r = 0; r < 8; ++r) {
                int mm = r + ((lane >> 4) * 8);
                S[mm * T_ + kt * 16 + nb] = c[r] * ATT_SCALE;
            }
        }
    }
    __syncthreads();

    // ---- Phase 2: causal softmax + write attentions ---------------------
    {
        int r = t >> 4, c0 = t & 15;
        int qrow = qb + r;

        float mx = -3.0e38f;
        for (int j = 0; j < T_ / 16; ++j) {
            int col = c0 + 16 * j;
            if (col <= qrow) {
                float v = S[r * T_ + col];
                mx = v > mx ? v : mx;
            }
        }
        red[r * 16 + c0] = mx;
        __syncthreads();
        if (c0 == 0) {
            float m2 = red[r * 16];
            for (int i = 1; i < 16; ++i) {
                float v = red[r * 16 + i];
                m2 = v > m2 ? v : m2;
            }
            rowv[r] = m2;
        }
        __syncthreads();

        float rm = rowv[r];
        float s = 0.0f;
        for (int j = 0; j < T_ / 16; ++j) {
            int col = c0 + 16 * j;
            float v = (col <= qrow) ? __expf(S[r * T_ + col] - rm) : 0.0f;
            S[r * T_ + col] = v;
            s += v;
        }
        __syncthreads();
        red[r * 16 + c0] = s;
        __syncthreads();
        if (c0 == 0) {
            float s2 = 0.0f;
            for (int i = 0; i < 16; ++i) s2 += red[r * 16 + i];
            rowv[r] = s2;
        }
        __syncthreads();

        float inv = 1.0f / rowv[r];
        float* arow = att + (bh * T_ + qrow) * (size_t)T_;
        for (int j = 0; j < T_ / 16; ++j) {
            int col = c0 + 16 * j;
            float p = S[r * T_ + col] * inv;
            S[r * T_ + col] = p;
            arow[col] = p;
        }
    }
    __syncthreads();

    // ---- Phase 3: Y = P V ------------------------------------------------
    if (wave < 4) {
        int nt = wave;
        v8f acc = {};
        const u16* vrow = Vt + (bh * D_ + nt * 16 + nb) * T_;
        for (int ks = 0; ks < T_; ks += 32) {
            Frag a;
#pragma unroll
            for (int e = 0; e < 8; ++e) {
                a.u[e]     = f2bf(S[mrow * T_ + ks + b0 + e]);
                a.u[8 + e] = f2bf(S[mrow * T_ + ks + b0 + 16 + e]);
            }
            Frag bf;
            const u16* vp = vrow + ks + kb;
            bf.q[0] = *(const u32x4*)vp;
            bf.q[1] = *(const u32x4*)(vp + 16);
            acc = wmma_bf16(a.v, bf.v, acc);
        }
#pragma unroll
        for (int r = 0; r < 8; ++r) {
            int mm = r + ((lane >> 4) * 8);
            Ybf[((size_t)(b * T_ + qb + mm)) * E_ + h * D_ + nt * 16 + nb] =
                f2bf(acc[r]);
        }
    }
}

// ---------------------------------------------------------------------------
// Cross-entropy: per-row NLL (V=256, one block/row), then ordered reduce
// ---------------------------------------------------------------------------
__global__ void __launch_bounds__(256) k_loss_rows(const float* __restrict__ logits,
                                                   const int* __restrict__ targets,
                                                   float* __restrict__ nll,
                                                   float* __restrict__ valid) {
    __shared__ float red[256];
    int row = blockIdx.x, t = threadIdx.x;
    float l = logits[(size_t)row * V_ + t];

    red[t] = l;
    __syncthreads();
    for (int o = 128; o > 0; o >>= 1) {
        if (t < o) red[t] = red[t] > red[t + o] ? red[t] : red[t + o];
        __syncthreads();
    }
    float mx = red[0];
    __syncthreads();

    red[t] = __expf(l - mx);
    __syncthreads();
    for (int o = 128; o > 0; o >>= 1) {
        if (t < o) red[t] += red[t + o];
        __syncthreads();
    }
    if (t == 0) {
        float lse = logf(red[0]) + mx;
        int tg = targets[row];
        if (tg >= 0) {
            nll[row]   = lse - logits[(size_t)row * V_ + tg];
            valid[row] = 1.0f;
        } else {
            nll[row]   = 0.0f;
            valid[row] = 0.0f;
        }
    }
}

__global__ void __launch_bounds__(256) k_loss_reduce(const float* __restrict__ nll,
                                                     const float* __restrict__ valid,
                                                     float* __restrict__ loss) {
    __shared__ float rs[256], rc[256];
    int t = threadIdx.x;
    float s = 0.0f, c = 0.0f;
    for (int i = t; i < BT_; i += 256) {
        s += nll[i];
        c += valid[i];
    }
    rs[t] = s; rc[t] = c;
    __syncthreads();
    for (int o = 128; o > 0; o >>= 1) {
        if (t < o) { rs[t] += rs[t + o]; rc[t] += rc[t + o]; }
        __syncthreads();
    }
    if (t == 0) {
        float cnt = rc[0];
        loss[0] = rs[0] / (cnt > 1.0f ? cnt : 1.0f);
    }
}

// ---------------------------------------------------------------------------
// Host-side orchestration
// ---------------------------------------------------------------------------
extern "C" void kernel_launch(void* const* d_in, const int* in_sizes, int n_in,
                              void* d_out, int out_size, void* d_ws, size_t ws_size,
                              hipStream_t stream) {
    (void)in_sizes; (void)n_in; (void)out_size; (void)ws_size;

    const int*   idx     = (const int*)d_in[0];
    const int*   targets = (const int*)d_in[1];
    const float* wte     = (const float*)d_in[2];
    const float* wpe     = (const float*)d_in[3];
    const float* ln1_g   = (const float*)d_in[4];
    const float* ln1_b   = (const float*)d_in[5];
    const float* attn_w  = (const float*)d_in[6];
    const float* attn_b  = (const float*)d_in[7];
    const float* proj_w  = (const float*)d_in[8];
    const float* proj_b  = (const float*)d_in[9];
    const float* ln2_g   = (const float*)d_in[10];
    const float* ln2_b   = (const float*)d_in[11];
    const float* fc_w    = (const float*)d_in[12];
    const float* fc_b    = (const float*)d_in[13];
    const float* fc2_w   = (const float*)d_in[14];
    const float* fc2_b   = (const float*)d_in[15];
    const float* lnf_g   = (const float*)d_in[16];
    const float* lnf_b   = (const float*)d_in[17];
    const float* head_w  = (const float*)d_in[18];

    float* out      = (float*)d_out;
    float* logits   = out;                              // [B,T,V]
    float* loss     = out + (size_t)BT_ * V_;           // [1]
    float* att_base = loss + 1;                         // [L,B,H,T,T]

    // ---- workspace carve ----
    size_t off = 0;
    auto alloc = [&](size_t bytes) -> char* {
        off = (off + 255) & ~(size_t)255;
        char* p = (char*)d_ws + off;
        off += bytes;
        return p;
    };
    float* x_f32   = (float*)alloc((size_t)BT_ * E_ * 4);
    float* qkv_f32 = (float*)alloc((size_t)BT_ * E3_ * 4);
    u16*   h_bf    = (u16*)alloc((size_t)BT_ * E_ * 2);
    u16*   mid_bf  = (u16*)alloc((size_t)BT_ * E4_ * 2);
    u16*   y_bf    = (u16*)alloc((size_t)BT_ * E_ * 2);
    u16*   q_bf    = (u16*)alloc((size_t)BT_ * E_ * 2);
    u16*   k_bf    = (u16*)alloc((size_t)BT_ * E_ * 2);
    u16*   vt_bf   = (u16*)alloc((size_t)BT_ * E_ * 2);
    u16*   w_attn  = (u16*)alloc((size_t)L_ * E_ * E3_ * 2);
    u16*   w_proj  = (u16*)alloc((size_t)L_ * E_ * E_ * 2);
    u16*   w_fc    = (u16*)alloc((size_t)L_ * E_ * E4_ * 2);
    u16*   w_fc2   = (u16*)alloc((size_t)L_ * E4_ * E_ * 2);
    u16*   w_head  = (u16*)alloc((size_t)E_ * V_ * 2);
    float* nll_buf = (float*)alloc((size_t)BT_ * 4);
    float* val_buf = (float*)alloc((size_t)BT_ * 4);

    // ---- weight conversion (f32 -> bf16) ----
    auto cvt = [&](const float* src, u16* dst, long long n) {
        k_cvt_bf16<<<(unsigned)((n / 4 + 255) / 256), 256, 0, stream>>>(src, dst, n);
    };
    cvt(attn_w, w_attn, (long long)L_ * E_ * E3_);
    cvt(proj_w, w_proj, (long long)L_ * E_ * E_);
    cvt(fc_w,   w_fc,   (long long)L_ * E_ * E4_);
    cvt(fc2_w,  w_fc2,  (long long)L_ * E4_ * E_);
    cvt(head_w, w_head, (long long)E_ * V_);

    // ---- embedding ----
    k_embed<<<(BT_ * E_) / 256, 256, 0, stream>>>(idx, wte, wpe, x_f32);

    const unsigned attn_smem = 16 * T_ * 4 + 256 * 4 + 16 * 4;

    for (int l = 0; l < L_; ++l) {
        // LN1 -> bf16
        k_layernorm<<<BT_, 256, 0, stream>>>(x_f32, ln1_g + (size_t)l * E_,
                                             ln1_b + (size_t)l * E_, h_bf);
        // QKV GEMM: [4096,512] x [512,1536]
        k_gemm_bf16<<<dim3(E3_ / 64, BT_ / 128), 256, 0, stream>>>(
            h_bf, w_attn + (size_t)l * E_ * E3_, attn_b + (size_t)l * E3_,
            qkv_f32, nullptr, BT_, E3_, E_, 0);
        // split + per-head layouts
        k_qkv_split<<<(BT_ * E_) / 256, 256, 0, stream>>>(qkv_f32, q_bf, k_bf, vt_bf);
        // attention (writes softmax probs to output + Y bf16)
        k_attention<<<dim3(T_ / 16, H_, B_), 256, attn_smem, stream>>>(
            q_bf, k_bf, vt_bf,
            att_base + (size_t)l * B_ * H_ * T_ * T_, y_bf);
        // proj GEMM: [4096,512] x [512,512] -> x
        k_gemm_bf16<<<dim3(E_ / 64, BT_ / 128), 256, 0, stream>>>(
            y_bf, w_proj + (size_t)l * E_ * E_, proj_b + (size_t)l * E_,
            x_f32, nullptr, BT_, E_, E_, 0);
        // LN2 -> bf16
        k_layernorm<<<BT_, 256, 0, stream>>>(x_f32, ln2_g + (size_t)l * E_,
                                             ln2_b + (size_t)l * E_, h_bf);
        // FC GEMM + GELU: [4096,512] x [512,2048] -> bf16 mid
        k_gemm_bf16<<<dim3(E4_ / 64, BT_ / 128), 256, 0, stream>>>(
            h_bf, w_fc + (size_t)l * E_ * E4_, fc_b + (size_t)l * E4_,
            nullptr, mid_bf, BT_, E4_, E_, 1);
        // FC2 GEMM: [4096,2048] x [2048,512] -> x
        k_gemm_bf16<<<dim3(E_ / 64, BT_ / 128), 256, 0, stream>>>(
            mid_bf, w_fc2 + (size_t)l * E4_ * E_, fc2_b + (size_t)l * E_,
            x_f32, nullptr, BT_, E_, E4_, 0);
    }

    // final LN + head GEMM -> logits
    k_layernorm<<<BT_, 256, 0, stream>>>(x_f32, lnf_g, lnf_b, h_bf);
    k_gemm_bf16<<<dim3(V_ / 64, BT_ / 128), 256, 0, stream>>>(
        h_bf, w_head, nullptr, logits, nullptr, BT_, V_, E_, 0);

    // cross-entropy loss (deterministic two-stage)
    k_loss_rows<<<BT_, 256, 0, stream>>>(logits, targets, nll_buf, val_buf);
    k_loss_reduce<<<1, 256, 0, stream>>>(nll_buf, val_buf, loss);
}
